// Direction_Aware_MP_84817014161454
// MI455X (gfx1250) — matrix operand order repaired and checked by reading, not compiled, not run on hardware
//
#include <hip/hip_runtime.h>
#include <hip/hip_bf16.h>
#include <stdint.h>

typedef __attribute__((ext_vector_type(16))) __bf16 v16bf;
typedef __attribute__((ext_vector_type(8)))  float  v8f;
typedef __attribute__((ext_vector_type(2)))  float  v2f;

#define BB    32
#define NPI_  80
#define DD    1024
#define NOBJ  2560      // BB*NPI_
#define NEDGE 32768     // BB*1024

// ---------------------------------------------------------------------------
// f32 -> bf16 convert (row-major preserved)
// ---------------------------------------------------------------------------
__global__ void f2bf_kernel(const float* __restrict__ in, __bf16* __restrict__ out, int n) {
    int i = blockIdx.x * blockDim.x + threadIdx.x;
    if (i < n) out[i] = (__bf16)in[i];
}

// f32 (KxN) -> bf16 transposed (NxK), 32x32 LDS tile transpose, block (32,8)
__global__ void f2bf_t_kernel(const float* __restrict__ in, __bf16* __restrict__ out,
                              int K, int N) {
    __shared__ float tile[32][33];
    const int n0 = blockIdx.x * 32;
    const int k0 = blockIdx.y * 32;
    const int tx = threadIdx.x;
    const int ty = threadIdx.y;
    #pragma unroll
    for (int i = 0; i < 4; ++i)
        tile[ty + 8 * i][tx] = in[(size_t)(k0 + ty + 8 * i) * N + n0 + tx];
    __syncthreads();
    #pragma unroll
    for (int i = 0; i < 4; ++i)
        out[(size_t)(n0 + ty + 8 * i) * K + k0 + tx] = (__bf16)tile[tx][ty + 8 * i];
}

__global__ void zero_kernel(float* __restrict__ p, int n) {
    int i = blockIdx.x * blockDim.x + threadIdx.x;
    if (i < n) p[i] = 0.0f;
}

// ---------------------------------------------------------------------------
// bf16 WMMA GEMM: C(MxN) = act( A(MxK) @ B(KxN) + bias [+ residual] )
// B supplied TRANSPOSED (Bt: NxK) so both A and B fragments are contiguous.
// Tile: 64(M) x 128(N) per 256-thread block (8 waves, wave32).
// Double-buffered LDS staged with GLOBAL_LOAD_ASYNC_TO_LDS_B128 (ASYNCcnt),
// so the next K-tile's HBM fetch overlaps the current tile's WMMAs.
// mode: 0 = none, 1 = relu, 2 = relu(x + residual)
// ---------------------------------------------------------------------------
union FragBf {
    v16bf  v;
    __bf16 e[16];
    float4 f4[2];
};

__device__ __forceinline__ void async_b128(uint32_t lds_off, const void* gaddr) {
    asm volatile("global_load_async_to_lds_b128 %0, %1, off"
                 :: "v"(lds_off), "v"((uint64_t)(uintptr_t)gaddr) : "memory");
}
__device__ __forceinline__ void async_b128_o16(uint32_t lds_off, const void* gaddr) {
    asm volatile("global_load_async_to_lds_b128 %0, %1, off offset:16"
                 :: "v"(lds_off), "v"((uint64_t)(uintptr_t)gaddr) : "memory");
}
__device__ __forceinline__ void wait_async0() {
    asm volatile("s_wait_asynccnt 0x0" ::: "memory");
}

__global__ void gemm_bf16_kernel(const __bf16* __restrict__ A,
                                 const __bf16* __restrict__ Bt,
                                 const float*  __restrict__ bias,
                                 float*        __restrict__ C,
                                 const float*  __restrict__ residual,
                                 int M, int Nc, int K, int mode) {
    __shared__ __align__(16) __bf16 As [2][64  * 40];  // 64 rows (m)  x 32 halfs (k), pad->40
    __shared__ __align__(16) __bf16 BsT[2][128 * 40];  // 128 rows (n) x 32 halfs (k), pad->40

    const int tid  = threadIdx.x;
    const int lane = tid & 31;
    const int wave = tid >> 5;
    const int wy   = wave & 3;   // 0..3 : M sub-block
    const int wx   = wave >> 2;  // 0..1 : 64-col half

    const int rowBlock = blockIdx.y * 64;
    const int colBlock = blockIdx.x * 128;

    v8f acc[4] = {};

    const int l15  = lane & 15;
    const int kbA  = (lane < 16) ? 0 : 8;    // A frag: K groups {0-7,16-23} / {8-15,24-31}
    const int kbB  = (lane < 16) ? 0 : 8;    // B frag: K groups {0-7,16-23} / {8-15,24-31}
    const int mrow = wy * 16 + l15;

    // staging coordinates (constant across k-steps)
    const int ar = tid >> 2, acg = tid & 3;           // A: 64 rows x 4 groups of 8 halfs
    const int bn = tid >> 1, bhg = (tid & 1) * 16;    // B: 128 rows x 2 groups of 16 halfs

    const __bf16* aSrcBase = A  + (size_t)(rowBlock + ar) * K + acg * 8;
    const __bf16* bSrcBase = Bt + (size_t)(colBlock + bn) * K + bhg;

    // per-thread LDS destination byte-offsets for both buffers
    uint32_t aDst[2], bDst[2];
    #pragma unroll
    for (int p = 0; p < 2; ++p) {
        aDst[p] = (uint32_t)(uintptr_t)(&As [p][ar * 40 + acg * 8]);
        bDst[p] = (uint32_t)(uintptr_t)(&BsT[p][bn * 40 + bhg]);
    }

    const int nk = K >> 5;   // K / 32 steps

    // prologue: async-stage tile 0 into buffer 0
    async_b128(aDst[0], aSrcBase);
    async_b128(bDst[0], bSrcBase);          // first 16B of the 32B B-row
    async_b128_o16(bDst[0], bSrcBase);      // offset:16 applies to BOTH lds+global addr

    for (int ks = 0; ks < nk; ++ks) {
        const int cur = ks & 1;
        wait_async0();        // this wave's async writes into buf[cur] have landed
        __syncthreads();      // all waves' writes visible; buf[1-cur] readers drained

        if (ks + 1 < nk) {    // overlap next tile fetch with this tile's WMMAs
            const int k0n = (ks + 1) << 5;
            async_b128(aDst[1 - cur], aSrcBase + k0n);
            async_b128(bDst[1 - cur], bSrcBase + k0n);
            async_b128_o16(bDst[1 - cur], bSrcBase + k0n);
        }

        // ---- A fragment: two b128 LDS loads ----
        FragBf a;
        a.f4[0] = *(const float4*)(&As[cur][mrow * 40 + kbA]);       // K = kbA .. kbA+7
        a.f4[1] = *(const float4*)(&As[cur][mrow * 40 + kbA + 16]);  // K = kbA+16 .. kbA+23

        // ---- 4 B fragments (two b128 LDS loads each) + WMMA ----
        #pragma unroll
        for (int t = 0; t < 4; ++t) {
            const int n = wx * 64 + t * 16 + l15;
            FragBf b;
            b.f4[0] = *(const float4*)(&BsT[cur][n * 40 + kbB]);       // K = kbB .. kbB+7
            b.f4[1] = *(const float4*)(&BsT[cur][n * 40 + kbB + 16]);  // K = kbB+16 .. kbB+23
            acc[t] = __builtin_amdgcn_wmma_f32_16x16x32_bf16(
                false, a.v, false, b.v, (short)0, acc[t], false, false);
        }
    }

    // ---- epilogue: C layout VGPR r -> M = r (lanes 0-15) / 8+r (lanes 16-31) ----
    const int hi = (lane >> 4) * 8;
    #pragma unroll
    for (int t = 0; t < 4; ++t) {
        const int n  = colBlock + wx * 64 + t * 16 + l15;
        const float bv = bias[n];
        #pragma unroll
        for (int r = 0; r < 8; ++r) {
            const int m = rowBlock + wy * 16 + r + hi;
            float v = acc[t][r] + bv;
            if (mode == 1) v = fmaxf(v, 0.0f);
            else if (mode == 2) v = fmaxf(v + residual[(size_t)m * Nc + n], 0.0f);
            C[(size_t)m * Nc + n] = v;
        }
    }
}

// ---------------------------------------------------------------------------
// Edge scoring: atten_f[e] = sum_d s[gi,d]*o[gj,d]*phr[e,d]*Wp[d] + bp
// scatter-add into Araw[b, li, lj]. One 256-thread block per edge.
// ---------------------------------------------------------------------------
__global__ void atten_kernel(const float* __restrict__ sF,
                             const float* __restrict__ oF,
                             const float* __restrict__ phr,
                             const float* __restrict__ Wp,
                             const float* __restrict__ bp,
                             const int*   __restrict__ rel,
                             float*       __restrict__ Araw) {
    __shared__ float sred[256];
    const int e  = blockIdx.x;
    const int b  = rel[e * 3 + 0];
    const int gi = rel[e * 3 + 1];
    const int gj = rel[e * 3 + 2];

    const float4* sr   = (const float4*)(sF + (size_t)gi * DD);
    const float4* orow = (const float4*)(oF + (size_t)gj * DD);
    const float4* ph   = (const float4*)(phr + (size_t)e * DD);
    const float4* wp   = (const float4*)Wp;

    const int t = threadIdx.x;            // 256 threads * 4 floats = 1024
    float4 sv = sr[t], ov = orow[t], pv = ph[t], wv = wp[t];
    float partial = sv.x * ov.x * pv.x * wv.x
                  + sv.y * ov.y * pv.y * wv.y
                  + sv.z * ov.z * pv.z * wv.z
                  + sv.w * ov.w * pv.w * wv.w;
    sred[t] = partial;
    __syncthreads();
    for (int s = 128; s > 0; s >>= 1) {
        if (t < s) sred[t] += sred[t + s];
        __syncthreads();
    }
    if (t == 0) {
        const int li = gi - b * NPI_;
        const int lj = gj - b * NPI_;
        atomicAdd(&Araw[b * NPI_ * NPI_ + li * NPI_ + lj], sred[0] + bp[0]);
    }
}

// ---------------------------------------------------------------------------
// A = sigmoid(Araw); zero diagonal; row sums S[b,i] = sum_j A[b,i,j]
// ---------------------------------------------------------------------------
__global__ void sig_rowsum_kernel(const float* __restrict__ Araw,
                                  float* __restrict__ Asig,
                                  float* __restrict__ S) {
    __shared__ float sred[128];
    const int idx = blockIdx.x;       // b*80 + i
    const int b = idx / NPI_;
    const int i = idx % NPI_;
    const int j = threadIdx.x;
    float v = 0.0f;
    if (j < NPI_) {
        float x = Araw[b * NPI_ * NPI_ + i * NPI_ + j];
        v = 1.0f / (1.0f + __expf(-x));
        if (j == i) v = 0.0f;
        Asig[b * NPI_ * NPI_ + i * NPI_ + j] = v;
    }
    sred[j] = v;
    __syncthreads();
    for (int s = 64; s > 0; s >>= 1) {
        if (j < s) sred[j] += sred[j + s];
        __syncthreads();
    }
    if (j == 0) S[idx] = sred[0];
}

// A[b,i,j] = Asig[b,i,j] / S[b,j]   (denominator indexed by COLUMN j)
__global__ void colnorm_kernel(const float* __restrict__ Asig,
                               const float* __restrict__ S,
                               float* __restrict__ An, int n) {
    int idx = blockIdx.x * blockDim.x + threadIdx.x;
    if (idx >= n) return;
    int j    = idx % NPI_;
    int rest = idx / NPI_;
    int b    = rest / NPI_;
    An[idx] = Asig[idx] / S[b * NPI_ + j];
}

// ---------------------------------------------------------------------------
// ctx half: Ctx[b,m, outoff+n] = (transA ? A^T : A) @ c   via f32 WMMA 16x16x4
// One wave per 16x16 output tile. A: (B,80,80) f32, c: (B,80,512) f32.
// ---------------------------------------------------------------------------
__global__ void ctx_wmma_f32_kernel(const float* __restrict__ An,
                                    const float* __restrict__ Cf,
                                    float* __restrict__ Ctx,
                                    int transA, int outoff) {
    const int lane = threadIdx.x;     // 32 threads = 1 wave
    const int b  = blockIdx.z;
    const int m0 = blockIdx.y * 16;   // 80 = 5*16
    const int n0 = blockIdx.x * 16;   // 512 = 32*16

    const float* Ab = An + (size_t)b * NPI_ * NPI_;
    const float* cb = Cf + (size_t)b * NPI_ * 512;

    const int l15  = lane & 15;
    const int koff = (lane < 16) ? 0 : 2;  // A 16x4 f32: lanes0-15 K{0,1}, lanes16-31 K{2,3}
    const int m = m0 + l15;
    const int n = n0 + l15;

    v8f acc = {};
    for (int k0 = 0; k0 < NPI_; k0 += 4) {
        v2f av, bv;
        if (!transA) {
            av[0] = Ab[m * NPI_ + k0 + koff];
            av[1] = Ab[m * NPI_ + k0 + koff + 1];
        } else {
            av[0] = Ab[(k0 + koff) * NPI_ + m];
            av[1] = Ab[(k0 + koff + 1) * NPI_ + m];
        }
        bv[0] = cb[(k0 + koff) * 512 + n];
        bv[1] = cb[(k0 + koff + 1) * 512 + n];
        acc = __builtin_amdgcn_wmma_f32_16x16x4_f32(
            false, av, false, bv, (short)0, acc, false, false);
    }
    const int hi = (lane >> 4) * 8;
    #pragma unroll
    for (int r = 0; r < 8; ++r) {
        const int mm = m0 + r + hi;
        Ctx[((size_t)(b * NPI_ + mm)) * DD + outoff + n] = acc[r];
    }
}

// ---------------------------------------------------------------------------
// LayerNorm(256) + ReLU -> bf16. One 256-thread block per row.
// ---------------------------------------------------------------------------
__global__ void ln_relu_kernel(const float* __restrict__ Hpre,
                               const float* __restrict__ gamma,
                               const float* __restrict__ beta,
                               __bf16* __restrict__ Hh) {
    __shared__ float sb[256];
    const int row = blockIdx.x;
    const int t   = threadIdx.x;
    const float x = Hpre[(size_t)row * 256 + t];
    sb[t] = x;
    __syncthreads();
    for (int s = 128; s > 0; s >>= 1) {
        if (t < s) sb[t] += sb[t + s];
        __syncthreads();
    }
    const float mean = sb[0] * (1.0f / 256.0f);
    __syncthreads();
    const float d = x - mean;
    sb[t] = d * d;
    __syncthreads();
    for (int s = 128; s > 0; s >>= 1) {
        if (t < s) sb[t] += sb[t + s];
        __syncthreads();
    }
    const float var = sb[0] * (1.0f / 256.0f);
    float y = d * rsqrtf(var + 1e-5f) * gamma[t] + beta[t];
    y = fmaxf(y, 0.0f);
    Hh[(size_t)row * 256 + t] = (__bf16)y;
}

// ---------------------------------------------------------------------------
// Host-side launch sequence
// ---------------------------------------------------------------------------
static inline size_t ws_alloc(size_t& off, size_t bytes) {
    size_t o = off;
    off += (bytes + 255) & ~(size_t)255;
    return o;
}

extern "C" void kernel_launch(void* const* d_in, const int* in_sizes, int n_in,
                              void* d_out, int out_size, void* d_ws, size_t ws_size,
                              hipStream_t stream) {
    const float* obj   = (const float*)d_in[0];
    const float* phr   = (const float*)d_in[1];
    // d_in[2] = im_inds (unused; batch recovered from rel_inds)
    const int*   rel   = (const int*)d_in[3];
    const float* Ws    = (const float*)d_in[4];
    const float* bs    = (const float*)d_in[5];
    const float* Wo    = (const float*)d_in[6];
    const float* bo    = (const float*)d_in[7];
    const float* Wp    = (const float*)d_in[8];
    const float* bp    = (const float*)d_in[9];
    const float* Wc    = (const float*)d_in[10];
    const float* bc    = (const float*)d_in[11];
    const float* W1    = (const float*)d_in[12];
    const float* b1    = (const float*)d_in[13];
    const float* gamma = (const float*)d_in[14];
    const float* beta  = (const float*)d_in[15];
    const float* W2    = (const float*)d_in[16];
    const float* b2    = (const float*)d_in[17];
    float* out = (float*)d_out;

    char* ws = (char*)d_ws;
    size_t off = 0;
    __bf16* objh  = (__bf16*)(ws + ws_alloc(off, (size_t)NOBJ * DD * 2));
    __bf16* Wsht  = (__bf16*)(ws + ws_alloc(off, (size_t)DD * DD * 2));   // transposed
    __bf16* Woht  = (__bf16*)(ws + ws_alloc(off, (size_t)DD * DD * 2));   // transposed
    __bf16* Wcht  = (__bf16*)(ws + ws_alloc(off, (size_t)DD * 512 * 2));  // transposed
    __bf16* W1ht  = (__bf16*)(ws + ws_alloc(off, (size_t)DD * 256 * 2));  // transposed
    __bf16* W2ht  = (__bf16*)(ws + ws_alloc(off, (size_t)256 * DD * 2));  // transposed
    float*  sfeat = (float*)(ws + ws_alloc(off, (size_t)NOBJ * DD * 4));
    float*  ofeat = (float*)(ws + ws_alloc(off, (size_t)NOBJ * DD * 4));
    float*  cfeat = (float*)(ws + ws_alloc(off, (size_t)NOBJ * 512 * 4));
    float*  Araw  = (float*)(ws + ws_alloc(off, (size_t)BB * NPI_ * NPI_ * 4));
    float*  Asig  = (float*)(ws + ws_alloc(off, (size_t)BB * NPI_ * NPI_ * 4));
    float*  Srow  = (float*)(ws + ws_alloc(off, (size_t)BB * NPI_ * 4));
    float*  Anorm = (float*)(ws + ws_alloc(off, (size_t)BB * NPI_ * NPI_ * 4));
    float*  ctx   = (float*)(ws + ws_alloc(off, (size_t)NOBJ * DD * 4));
    __bf16* ctxh  = (__bf16*)(ws + ws_alloc(off, (size_t)NOBJ * DD * 2));
    float*  hpre  = (float*)(ws + ws_alloc(off, (size_t)NOBJ * 256 * 4));
    __bf16* hh    = (__bf16*)(ws + ws_alloc(off, (size_t)NOBJ * 256 * 2));
    (void)ws_size; (void)in_sizes; (void)n_in; (void)out_size;

    // ---- converts (weights transposed so GEMM B-fragments are contiguous) ----
    f2bf_kernel<<<(NOBJ * DD + 255) / 256, 256, 0, stream>>>(obj, objh, NOBJ * DD);
    f2bf_t_kernel<<<dim3(DD / 32, DD / 32),  dim3(32, 8), 0, stream>>>(Ws, Wsht, DD, DD);
    f2bf_t_kernel<<<dim3(DD / 32, DD / 32),  dim3(32, 8), 0, stream>>>(Wo, Woht, DD, DD);
    f2bf_t_kernel<<<dim3(512 / 32, DD / 32), dim3(32, 8), 0, stream>>>(Wc, Wcht, DD, 512);
    f2bf_t_kernel<<<dim3(256 / 32, DD / 32), dim3(32, 8), 0, stream>>>(W1, W1ht, DD, 256);
    f2bf_t_kernel<<<dim3(DD / 32, 256 / 32), dim3(32, 8), 0, stream>>>(W2, W2ht, 256, DD);

    // ---- big GEMMs (bf16 WMMA, async-LDS double buffered) ----
    gemm_bf16_kernel<<<dim3(DD / 128, NOBJ / 64), 256, 0, stream>>>(
        objh, Wsht, bs, sfeat, nullptr, NOBJ, DD, DD, 0);
    gemm_bf16_kernel<<<dim3(DD / 128, NOBJ / 64), 256, 0, stream>>>(
        objh, Woht, bo, ofeat, nullptr, NOBJ, DD, DD, 0);
    gemm_bf16_kernel<<<dim3(512 / 128, NOBJ / 64), 256, 0, stream>>>(
        objh, Wcht, bc, cfeat, nullptr, NOBJ, 512, DD, 1);

    // ---- attention logits ----
    zero_kernel<<<(BB * NPI_ * NPI_ + 255) / 256, 256, 0, stream>>>(Araw, BB * NPI_ * NPI_);
    atten_kernel<<<NEDGE, 256, 0, stream>>>(sfeat, ofeat, phr, Wp, bp, rel, Araw);

    // ---- normalize ----
    sig_rowsum_kernel<<<BB * NPI_, 128, 0, stream>>>(Araw, Asig, Srow);
    colnorm_kernel<<<(BB * NPI_ * NPI_ + 255) / 256, 256, 0, stream>>>(
        Asig, Srow, Anorm, BB * NPI_ * NPI_);

    // ---- context aggregation (f32 WMMA 16x16x4) ----
    ctx_wmma_f32_kernel<<<dim3(512 / 16, NPI_ / 16, BB), 32, 0, stream>>>(
        Anorm, cfeat, ctx, 0, 0);
    ctx_wmma_f32_kernel<<<dim3(512 / 16, NPI_ / 16, BB), 32, 0, stream>>>(
        Anorm, cfeat, ctx, 1, 512);

    // ---- MLP head ----
    f2bf_kernel<<<(NOBJ * DD + 255) / 256, 256, 0, stream>>>(ctx, ctxh, NOBJ * DD);
    gemm_bf16_kernel<<<dim3(256 / 128, NOBJ / 64), 256, 0, stream>>>(
        ctxh, W1ht, b1, hpre, nullptr, NOBJ, 256, DD, 0);
    ln_relu_kernel<<<NOBJ, 256, 0, stream>>>(hpre, gamma, beta, hh);
    gemm_bf16_kernel<<<dim3(DD / 128, NOBJ / 64), 256, 0, stream>>>(
        hh, W2ht, b2, out, obj, NOBJ, DD, 256, 2);
}
